// ResNetStages_79216376807742
// MI455X (gfx1250) — compile-verified, matrix operand order checked by reading, not run on hardware
//
#include <hip/hip_runtime.h>
#include <hip/hip_bf16.h>
#include <math.h>

// ---------------------------------------------------------------------------
// MI455X (gfx1250) ResNet-stage + DCT-quant pipeline.
//   * convs: implicit GEMM, NHWC activations, [Cout][tap][Cin] weights,
//     32x32 C-tile per wave => 4x v_wmma_f32_16x16x32_bf16 per K-step,
//     float4 (b128) fragment loads, b128 stores.
//   * DCT/IDCT: f32 matrix pipe (v_wmma_f32_16x16x4_f32), LDS-tiled,
//     zero-padded 56->64 / 28->32, map payload staged with
//     GLOBAL_LOAD_ASYNC_TO_LDS_B32 (+ s_wait_asynccnt) — no VGPR round-trip.
//   * BN: two-stage deterministic batch statistics (biased variance).
// ---------------------------------------------------------------------------

typedef __attribute__((ext_vector_type(16))) __bf16 v16bf;
typedef __attribute__((ext_vector_type(8)))  float  v8f;
typedef __attribute__((ext_vector_type(2)))  float  v2f;

#define BATCH  16
#define BN_EPS 1e-5f
static __device__ __constant__ float PI_F = 3.14159265358979323846f;

__device__ __forceinline__ v8f wmma_bf16(v16bf a, v16bf b, v8f c) {
    return __builtin_amdgcn_wmma_f32_16x16x32_bf16(false, a, false, b,
                                                   (short)0, c, false, false);
}
__device__ __forceinline__ v8f wmma_f32x4(v2f a, v2f b, v8f c) {
    return __builtin_amdgcn_wmma_f32_16x16x4_f32(false, a, false, b,
                                                 (short)0, c, false, false);
}

// CDNA5 async global->LDS copy (ASYNCcnt-tracked), inline asm for portability
// across toolchains (builtin arity differs between ROCm 7.2 and clang-23).
__device__ __forceinline__ void async_ld_b32(void* lds_dst, const void* gsrc) {
    asm volatile("global_load_async_to_lds_b32 %0, %1, off"
                 :: "v"((unsigned)(size_t)lds_dst),
                    "v"((unsigned long long)(size_t)gsrc)
                 : "memory");
}
__device__ __forceinline__ void wait_async0() {
    asm volatile("s_wait_asynccnt 0" ::: "memory");
}

// load 8 consecutive f32 -> bf16 fragment elements [BASE..BASE+7]
template<int BASE>
__device__ __forceinline__ void load8_bf16(const float* __restrict__ p, v16bf& f) {
    const float4 q0 = *(const float4*)p;
    const float4 q1 = *(const float4*)(p + 4);
    f[BASE + 0] = (__bf16)q0.x; f[BASE + 1] = (__bf16)q0.y;
    f[BASE + 2] = (__bf16)q0.z; f[BASE + 3] = (__bf16)q0.w;
    f[BASE + 4] = (__bf16)q1.x; f[BASE + 5] = (__bf16)q1.y;
    f[BASE + 6] = (__bf16)q1.z; f[BASE + 7] = (__bf16)q1.w;
}
template<int BASE>
__device__ __forceinline__ void zero8_bf16(v16bf& f) {
    f[BASE + 0] = (__bf16)0.f; f[BASE + 1] = (__bf16)0.f;
    f[BASE + 2] = (__bf16)0.f; f[BASE + 3] = (__bf16)0.f;
    f[BASE + 4] = (__bf16)0.f; f[BASE + 5] = (__bf16)0.f;
    f[BASE + 6] = (__bf16)0.f; f[BASE + 7] = (__bf16)0.f;
}

// ---------------------------------------------------------------------------
// Implicit-GEMM conv, NHWC in / NHWC out, weights pre-transposed to
// wt[m][tap][ci].  K = tap*CIN + ci (CIN multiple of 32 => one tap per chunk).
// One wave computes a 32x32 output tile: acc[mTile][nTile], 4 WMMA per K-step.
// ---------------------------------------------------------------------------
template<int CIN, int COUT, int H, int W, int STRIDE, int KSZ>
__global__ __launch_bounds__(256)
void conv_wmma_kernel(const float* __restrict__ x, const float* __restrict__ wt,
                      float* __restrict__ out)
{
    constexpr int OH   = H / STRIDE;
    constexpr int OW   = W / STRIDE;
    constexpr int PAD  = (KSZ == 3) ? 1 : 0;
    constexpr int TAPS = KSZ * KSZ;
    constexpr int NTOT = BATCH * OH * OW;

    const int lane   = threadIdx.x & 31;
    const int wave   = threadIdx.x >> 5;
    const int nBase  = (blockIdx.x * 8 + wave) * 32;
    if (nBase >= NTOT) return;                    // wave-uniform
    const int mBase  = blockIdx.y * 32;

    const int halfSel = lane >> 4;
    const int ks      = halfSel * 8;
    const int lm      = lane & 15;

    int bb[2], ohv[2], owv[2];
#pragma unroll
    for (int j = 0; j < 2; ++j) {
        const int n = nBase + j * 16 + lm;
        bb[j] = n / (OH * OW);
        const int os = n % (OH * OW);
        ohv[j] = os / OW; owv[j] = os % OW;
    }

    v8f acc[2][2] = {};

    for (int tap = 0; tap < TAPS; ++tap) {
        const int kh = tap / KSZ, kw = tap % KSZ;
        const float* bptr[2];
#pragma unroll
        for (int j = 0; j < 2; ++j) {
            const int ih = ohv[j] * STRIDE - PAD + kh;
            const int iw = owv[j] * STRIDE - PAD + kw;
            bptr[j] = (ih >= 0 && ih < H && iw >= 0 && iw < W)
                        ? &x[(((size_t)bb[j] * H + ih) * W + iw) * CIN] : nullptr;
        }
        if (bptr[0]) __builtin_prefetch(bptr[0], 0, 3);   // global_prefetch_b8

        for (int cb = 0; cb < CIN; cb += 32) {
            v16bf afrag[2], bfrag[2];
#pragma unroll
            for (int t = 0; t < 2; ++t) {
                const float* ap =
                    &wt[(((size_t)(mBase + t * 16 + lm)) * TAPS + tap) * CIN + cb + ks];
                load8_bf16<0>(ap,      afrag[t]);
                load8_bf16<8>(ap + 16, afrag[t]);
            }
#pragma unroll
            for (int j = 0; j < 2; ++j) {
                if (bptr[j]) {
                    const float* bp = bptr[j] + cb + ks;
                    load8_bf16<0>(bp,      bfrag[j]);
                    load8_bf16<8>(bp + 16, bfrag[j]);
                } else {
                    zero8_bf16<0>(bfrag[j]);
                    zero8_bf16<8>(bfrag[j]);
                }
            }
#pragma unroll
            for (int t = 0; t < 2; ++t)
#pragma unroll
                for (int j = 0; j < 2; ++j)
                    acc[t][j] = wmma_bf16(afrag[t], bfrag[j], acc[t][j]);
        }
    }

#pragma unroll
    for (int t = 0; t < 2; ++t)
#pragma unroll
        for (int j = 0; j < 2; ++j) {
            float* op = &out[(((size_t)bb[j] * OH + ohv[j]) * OW + owv[j]) * COUT
                             + mBase + t * 16 + 8 * halfSel];
            *(float4*)op       = make_float4(acc[t][j][0], acc[t][j][1],
                                             acc[t][j][2], acc[t][j][3]);
            *(float4*)(op + 4) = make_float4(acc[t][j][4], acc[t][j][5],
                                             acc[t][j][6], acc[t][j][7]);
        }
}

// ---------------------------------------------------------------------------
// weight transpose OIHW -> [O][tap][I]
// ---------------------------------------------------------------------------
__global__ void wt_transform_kernel(const float* __restrict__ w, float* __restrict__ wt,
                                    int COUT, int CIN, int TAPS)
{
    const int i = blockIdx.x * blockDim.x + threadIdx.x;
    if (i >= COUT * CIN * TAPS) return;
    const int tap = i % TAPS;
    const int t   = i / TAPS;
    const int ci  = t % CIN;
    const int m   = t / CIN;
    wt[((size_t)m * TAPS + tap) * CIN + ci] = w[i];
}

// ---------------------------------------------------------------------------
// NCHW -> NHWC copy (for the network input)
// ---------------------------------------------------------------------------
__global__ void to_nhwc_kernel(const float* __restrict__ src, float* __restrict__ dst,
                               int C, int HW, int total)
{
    const int i = blockIdx.x * blockDim.x + threadIdx.x;
    if (i >= total) return;
    const int hw = i % HW;
    const int t  = i / HW;
    const int c  = t % C;
    const int b  = t / C;
    dst[((size_t)b * HW + hw) * C + c] = src[i];
}

// ---------------------------------------------------------------------------
// Deterministic two-stage BN statistics over NHWC rows.
// ---------------------------------------------------------------------------
template<int C>
__global__ __launch_bounds__(256)
void bn_stats_part(const float* __restrict__ x, int R,
                   float* __restrict__ pm, float* __restrict__ pq)
{
    constexpr int SUBS = 256 / C;
    const int c   = threadIdx.x % C;
    const int sub = threadIdx.x / C;
    const int rowsPer = (R + gridDim.x - 1) / gridDim.x;
    const int r0 = blockIdx.x * rowsPer;
    int r1 = r0 + rowsPer; if (r1 > R) r1 = R;

    float s = 0.f, sq = 0.f;
    for (int r = r0 + sub; r < r1; r += SUBS) {
        const float v = x[(size_t)r * C + c];
        s += v; sq += v * v;
    }
    __shared__ float ss[256], ssq[256];
    ss[threadIdx.x] = s; ssq[threadIdx.x] = sq;
    __syncthreads();
    if (sub == 0) {
        for (int u = 1; u < SUBS; ++u) { s += ss[u * C + c]; sq += ssq[u * C + c]; }
        pm[blockIdx.x * C + c] = s;
        pq[blockIdx.x * C + c] = sq;
    }
}

__global__ void bn_stats_final(const float* __restrict__ pm, const float* __restrict__ pq,
                               int C, int nPart, int nElems,
                               float* __restrict__ mean, float* __restrict__ var)
{
    const int c = blockIdx.x * blockDim.x + threadIdx.x;
    if (c >= C) return;
    float s = 0.f, sq = 0.f;
    for (int p = 0; p < nPart; ++p) { s += pm[p * C + c]; sq += pq[p * C + c]; }
    const float m = s / nElems;
    mean[c] = m;
    var[c]  = sq / nElems - m * m;
}

// ---------------------------------------------------------------------------
// y = relu(bn(x))  (NHWC)
// ---------------------------------------------------------------------------
__global__ void bn_relu_kernel(const float* __restrict__ x, float* __restrict__ y,
                               const float* __restrict__ mean, const float* __restrict__ var,
                               const float* __restrict__ g, const float* __restrict__ b,
                               int C, int total)
{
    const int i = blockIdx.x * blockDim.x + threadIdx.x;
    if (i >= total) return;
    const int c = i % C;
    const float v = (x[i] - mean[c]) * rsqrtf(var[c] + BN_EPS) * g[c] + b[c];
    y[i] = fmaxf(v, 0.f);
}

// ---------------------------------------------------------------------------
// fm(NCHW) = relu(bn2(conv2) + identity), reads NHWC, identity optionally bn'd
// ---------------------------------------------------------------------------
__global__ void bn_add_relu_kernel(const float* __restrict__ conv2,
                                   const float* __restrict__ mean2, const float* __restrict__ var2,
                                   const float* __restrict__ g2, const float* __restrict__ b2,
                                   const float* __restrict__ idn,
                                   const float* __restrict__ meand, const float* __restrict__ vard,
                                   const float* __restrict__ gd, const float* __restrict__ bd,
                                   int down, float* __restrict__ fm,
                                   int C, int HW, int total)
{
    const int i = blockIdx.x * blockDim.x + threadIdx.x;
    if (i >= total) return;
    const int c   = i % C;
    const int row = i / C;
    const int b   = row / HW;
    const int hw  = row % HW;
    const float o = (conv2[i] - mean2[c]) * rsqrtf(var2[c] + BN_EPS) * g2[c] + b2[c];
    float id = idn[i];
    if (down)
        id = (id - meand[c]) * rsqrtf(vard[c] + BN_EPS) * gd[c] + bd[c];
    fm[((size_t)b * C + c) * HW + hw] = fmaxf(o + id, 0.f);
}

// ---------------------------------------------------------------------------
// DCT-II matrix and its analytic inverse
// ---------------------------------------------------------------------------
__global__ void dct_mats_kernel(float* __restrict__ D, float* __restrict__ Di, int N)
{
    const int i = blockIdx.x * blockDim.x + threadIdx.x;
    if (i >= N * N) return;
    const int k = i / N, n = i % N;
    const float ang = PI_F * (2.0f * n + 1.0f) * k / (2.0f * N);
    const float c = cosf(ang);
    D[k * N + n]  = 2.0f * c;
    Di[n * N + k] = (1.0f / N) * ((k == 0) ? 0.5f : 1.0f) * c;
}

// ---------------------------------------------------------------------------
// Per-map X = D*x*D^T on the f32 matrix pipe; Xq = clamp(rint(X/4),-128,127).
// Map + DCT matrix staged into LDS with async global->LDS copies.
// ---------------------------------------------------------------------------
template<int N, int NP>
__global__ __launch_bounds__(256)
void dct_quant_wmma(const float* __restrict__ fm, const float* __restrict__ D,
                    float* __restrict__ dq)
{
    constexpr int S  = NP + 1;
    constexpr int TI = NP / 16;
    __shared__ float xs[NP * S], ts[NP * S], Ds[NP * S];

    const float* src = fm + (size_t)blockIdx.x * N * N;
    for (int i = threadIdx.x; i < NP * NP; i += 256) {
        const int r = i / NP, c = i % NP;
        if (r < N && c < N) {
            async_ld_b32(&xs[r * S + c], &src[r * N + c]);
            async_ld_b32(&Ds[r * S + c], &D[r * N + c]);
        } else {
            xs[r * S + c] = 0.f;
            Ds[r * S + c] = 0.f;
        }
    }
    wait_async0();
    __syncthreads();

    const int lane = threadIdx.x & 31, wave = threadIdx.x >> 5;
    const int halfSel = lane >> 4, lm = lane & 15;
    const int koff = 2 * halfSel;

    // stage 1: ts = D * x
    for (int t = wave; t < TI * TI; t += 8) {
        const int mT = t / TI, nT = t % TI;
        v8f acc = {};
        for (int kb = 0; kb < NP; kb += 4) {
            v2f a, b;
            a[0] = Ds[(mT * 16 + lm) * S + kb + koff];
            a[1] = Ds[(mT * 16 + lm) * S + kb + koff + 1];
            b[0] = xs[(kb + koff) * S + nT * 16 + lm];
            b[1] = xs[(kb + koff + 1) * S + nT * 16 + lm];
            acc = wmma_f32x4(a, b, acc);
        }
#pragma unroll
        for (int r = 0; r < 8; ++r)
            ts[(mT * 16 + r + 8 * halfSel) * S + nT * 16 + lm] = acc[r];
    }
    __syncthreads();

    // stage 2: X = ts * D^T, quantize
    float* out = dq + (size_t)blockIdx.x * N * N;
    for (int t = wave; t < TI * TI; t += 8) {
        const int mT = t / TI, nT = t % TI;
        v8f acc = {};
        for (int kb = 0; kb < NP; kb += 4) {
            v2f a, b;
            a[0] = ts[(mT * 16 + lm) * S + kb + koff];
            a[1] = ts[(mT * 16 + lm) * S + kb + koff + 1];
            b[0] = Ds[(nT * 16 + lm) * S + kb + koff];
            b[1] = Ds[(nT * 16 + lm) * S + kb + koff + 1];
            acc = wmma_f32x4(a, b, acc);
        }
#pragma unroll
        for (int r = 0; r < 8; ++r) {
            const int k = mT * 16 + r + 8 * halfSel, l = nT * 16 + lm;
            if (k < N && l < N)
                out[k * N + l] = fminf(fmaxf(rintf(acc[r] * 0.25f), -128.f), 127.f);
        }
    }
}

// ---------------------------------------------------------------------------
// xr = Dinv * (4*Xq) * Dinv^T ; writes NHWC activations and/or NCHW output.
// Async raw copy of Xq; the *4 dequant scale is folded into the stage-1 store.
// ---------------------------------------------------------------------------
template<int N, int NP>
__global__ __launch_bounds__(256)
void idct_wmma(const float* __restrict__ Xq, const float* __restrict__ Di,
               float* __restrict__ outNHWC, int C, float* __restrict__ outNCHW)
{
    constexpr int S  = NP + 1;
    constexpr int TI = NP / 16;
    __shared__ float xs[NP * S], ts[NP * S], Ds[NP * S];

    const float* src = Xq + (size_t)blockIdx.x * N * N;
    for (int i = threadIdx.x; i < NP * NP; i += 256) {
        const int r = i / NP, c = i % NP;
        if (r < N && c < N) {
            async_ld_b32(&xs[r * S + c], &src[r * N + c]);
            async_ld_b32(&Ds[r * S + c], &Di[r * N + c]);
        } else {
            xs[r * S + c] = 0.f;
            Ds[r * S + c] = 0.f;
        }
    }
    wait_async0();
    __syncthreads();

    const int lane = threadIdx.x & 31, wave = threadIdx.x >> 5;
    const int halfSel = lane >> 4, lm = lane & 15;
    const int koff = 2 * halfSel;

    // stage 1: ts[m][w] = 4 * sum_n Xq[m][n] * Di[w][n]
    for (int t = wave; t < TI * TI; t += 8) {
        const int mT = t / TI, nT = t % TI;
        v8f acc = {};
        for (int kb = 0; kb < NP; kb += 4) {
            v2f a, b;
            a[0] = xs[(mT * 16 + lm) * S + kb + koff];
            a[1] = xs[(mT * 16 + lm) * S + kb + koff + 1];
            b[0] = Ds[(nT * 16 + lm) * S + kb + koff];
            b[1] = Ds[(nT * 16 + lm) * S + kb + koff + 1];
            acc = wmma_f32x4(a, b, acc);
        }
#pragma unroll
        for (int r = 0; r < 8; ++r)
            ts[(mT * 16 + r + 8 * halfSel) * S + nT * 16 + lm] = 4.0f * acc[r];
    }
    __syncthreads();

    // stage 2: xr[h][w] = sum_m Di[h][m] * ts[m][w]
    const int map = blockIdx.x;
    const int bN  = map / C, cN = map % C;
    for (int t = wave; t < TI * TI; t += 8) {
        const int mT = t / TI, nT = t % TI;
        v8f acc = {};
        for (int kb = 0; kb < NP; kb += 4) {
            v2f a, b;
            a[0] = Ds[(mT * 16 + lm) * S + kb + koff];
            a[1] = Ds[(mT * 16 + lm) * S + kb + koff + 1];
            b[0] = ts[(kb + koff) * S + nT * 16 + lm];
            b[1] = ts[(kb + koff + 1) * S + nT * 16 + lm];
            acc = wmma_f32x4(a, b, acc);
        }
#pragma unroll
        for (int r = 0; r < 8; ++r) {
            const int h = mT * 16 + r + 8 * halfSel, w = nT * 16 + lm;
            if (h < N && w < N) {
                const float v = acc[r];
                if (outNHWC) outNHWC[(((size_t)bN * N + h) * N + w) * C + cN] = v;
                if (outNCHW) outNCHW[(size_t)map * N * N + h * N + w] = v;
            }
        }
    }
}

// ---------------------------------------------------------------------------
extern "C" void kernel_launch(void* const* d_in, const int* in_sizes, int n_in,
                              void* d_out, int out_size, void* d_ws, size_t ws_size,
                              hipStream_t stream)
{
    (void)in_sizes; (void)n_in; (void)out_size; (void)ws_size;

    const float* x0  = (const float*)d_in[0];
    float*       out = (float*)d_out;
    float*       ws  = (float*)d_ws;

    const size_t S56 = (size_t)BATCH * 64 * 56 * 56;   // 3,211,264
    const size_t S28 = (size_t)BATCH * 128 * 28 * 28;  // 1,605,632
    const int R56 = BATCH * 56 * 56;                   // NHWC rows, C=64
    const int R28 = BATCH * 28 * 28;                   // NHWC rows, C=128
    const int NPART = 32;

    // workspace layout (floats)
    float* A    = ws;                    // NHWC activations between blocks
    float* Bb   = ws + S56;              // conv scratch (NHWC)
    float* Cact = ws + 2 * S56;          // post conv1-bn-relu (NHWC)
    float* Dn   = ws + 3 * S56;          // downsample conv out (NHWC)
    float* st   = ws + 3 * S56 + S28;
    float* mean1 = st,       *var1 = st + 128;
    float* mean2 = st + 256, *var2 = st + 384;
    float* meand = st + 512, *vard = st + 640;
    float* pm   = st + 1024;             // 32*128 partial sums
    float* pq   = pm + 4096;
    float* mats = pq + 4096;
    float* D56  = mats;
    float* Di56 = mats + 56 * 56;
    float* D28  = mats + 2 * 56 * 56;
    float* Di28 = mats + 2 * 56 * 56 + 28 * 28;
    float* Wt   = Di28 + 28 * 28;        // transposed weights, <=147456 floats

    // output layout: final x, fm[0..3], dct[0..3]  (all NCHW)
    const size_t fmsz[4] = {S56, S56, S28, S28};
    float* o_fm[4]; float* o_dq[4];
    size_t off = S28;
    for (int i = 0; i < 4; ++i) { o_fm[i] = out + off; off += fmsz[i]; }
    for (int i = 0; i < 4; ++i) { o_dq[i] = out + off; off += fmsz[i]; }

    struct BlkP { const float *w1,*g1,*b1,*w2,*g2,*b2,*wd,*gd,*bd; };
    BlkP P[4];
    int idx = 1;
    for (int i = 0; i < 4; ++i) {
        P[i].w1 = (const float*)d_in[idx++]; P[i].g1 = (const float*)d_in[idx++];
        P[i].b1 = (const float*)d_in[idx++]; P[i].w2 = (const float*)d_in[idx++];
        P[i].g2 = (const float*)d_in[idx++]; P[i].b2 = (const float*)d_in[idx++];
        if (i == 2) {
            P[i].wd = (const float*)d_in[idx++];
            P[i].gd = (const float*)d_in[idx++];
            P[i].bd = (const float*)d_in[idx++];
        } else { P[i].wd = P[i].gd = P[i].bd = nullptr; }
    }

    dct_mats_kernel<<<(56 * 56 + 255) / 256, 256, 0, stream>>>(D56, Di56, 56);
    dct_mats_kernel<<<(28 * 28 + 255) / 256, 256, 0, stream>>>(D28, Di28, 28);
    to_nhwc_kernel<<<((int)S56 + 255) / 256, 256, 0, stream>>>(x0, A, 64, 56 * 56, (int)S56);

    const dim3 gc56(196, 2);   // 50176/32/8 = 196 ; Cout 64  -> 2
    const dim3 gc28(49, 4);    // 12544/32/8 = 49  ; Cout 128 -> 4
    const int  t56 = (int)S56, t28 = (int)S28;
    const int  gp56 = (t56 + 255) / 256, gp28 = (t28 + 255) / 256;

    // ---- blocks 0 & 1 (64ch, 56x56, no downsample) -------------------------
    for (int blk = 0; blk < 2; ++blk) {
        wt_transform_kernel<<<(64*64*9 + 255)/256, 256, 0, stream>>>(P[blk].w1, Wt, 64, 64, 9);
        conv_wmma_kernel<64,64,56,56,1,3><<<gc56, 256, 0, stream>>>(A, Wt, Bb);
        bn_stats_part<64><<<NPART, 256, 0, stream>>>(Bb, R56, pm, pq);
        bn_stats_final<<<1, 128, 0, stream>>>(pm, pq, 64, NPART, R56, mean1, var1);
        bn_relu_kernel<<<gp56, 256, 0, stream>>>(Bb, Cact, mean1, var1,
                                                 P[blk].g1, P[blk].b1, 64, t56);
        wt_transform_kernel<<<(64*64*9 + 255)/256, 256, 0, stream>>>(P[blk].w2, Wt, 64, 64, 9);
        conv_wmma_kernel<64,64,56,56,1,3><<<gc56, 256, 0, stream>>>(Cact, Wt, Bb);
        bn_stats_part<64><<<NPART, 256, 0, stream>>>(Bb, R56, pm, pq);
        bn_stats_final<<<1, 128, 0, stream>>>(pm, pq, 64, NPART, R56, mean2, var2);
        bn_add_relu_kernel<<<gp56, 256, 0, stream>>>(Bb, mean2, var2, P[blk].g2, P[blk].b2,
                                                     A, nullptr, nullptr, nullptr, nullptr,
                                                     0, o_fm[blk], 64, 56 * 56, t56);
        dct_quant_wmma<56,64><<<BATCH * 64, 256, 0, stream>>>(o_fm[blk], D56, o_dq[blk]);
        idct_wmma<56,64><<<BATCH * 64, 256, 0, stream>>>(o_dq[blk], Di56, A, 64, nullptr);
    }

    // ---- block 2 (64->128, stride 2, downsample) ---------------------------
    {
        wt_transform_kernel<<<(128*64*9 + 255)/256, 256, 0, stream>>>(P[2].w1, Wt, 128, 64, 9);
        conv_wmma_kernel<64,128,56,56,2,3><<<gc28, 256, 0, stream>>>(A, Wt, Bb);
        bn_stats_part<128><<<NPART, 256, 0, stream>>>(Bb, R28, pm, pq);
        bn_stats_final<<<1, 128, 0, stream>>>(pm, pq, 128, NPART, R28, mean1, var1);
        bn_relu_kernel<<<gp28, 256, 0, stream>>>(Bb, Cact, mean1, var1,
                                                 P[2].g1, P[2].b1, 128, t28);
        wt_transform_kernel<<<(128*64 + 255)/256, 256, 0, stream>>>(P[2].wd, Wt, 128, 64, 1);
        conv_wmma_kernel<64,128,56,56,2,1><<<gc28, 256, 0, stream>>>(A, Wt, Dn);
        bn_stats_part<128><<<NPART, 256, 0, stream>>>(Dn, R28, pm, pq);
        bn_stats_final<<<1, 128, 0, stream>>>(pm, pq, 128, NPART, R28, meand, vard);
        wt_transform_kernel<<<(128*128*9 + 255)/256, 256, 0, stream>>>(P[2].w2, Wt, 128, 128, 9);
        conv_wmma_kernel<128,128,28,28,1,3><<<gc28, 256, 0, stream>>>(Cact, Wt, Bb);
        bn_stats_part<128><<<NPART, 256, 0, stream>>>(Bb, R28, pm, pq);
        bn_stats_final<<<1, 128, 0, stream>>>(pm, pq, 128, NPART, R28, mean2, var2);
        bn_add_relu_kernel<<<gp28, 256, 0, stream>>>(Bb, mean2, var2, P[2].g2, P[2].b2,
                                                     Dn, meand, vard, P[2].gd, P[2].bd,
                                                     1, o_fm[2], 128, 28 * 28, t28);
        dct_quant_wmma<28,32><<<BATCH * 128, 256, 0, stream>>>(o_fm[2], D28, o_dq[2]);
        idct_wmma<28,32><<<BATCH * 128, 256, 0, stream>>>(o_dq[2], Di28, A, 128, nullptr);
    }

    // ---- block 3 (128ch, 28x28, no downsample) -----------------------------
    {
        wt_transform_kernel<<<(128*128*9 + 255)/256, 256, 0, stream>>>(P[3].w1, Wt, 128, 128, 9);
        conv_wmma_kernel<128,128,28,28,1,3><<<gc28, 256, 0, stream>>>(A, Wt, Bb);
        bn_stats_part<128><<<NPART, 256, 0, stream>>>(Bb, R28, pm, pq);
        bn_stats_final<<<1, 128, 0, stream>>>(pm, pq, 128, NPART, R28, mean1, var1);
        bn_relu_kernel<<<gp28, 256, 0, stream>>>(Bb, Cact, mean1, var1,
                                                 P[3].g1, P[3].b1, 128, t28);
        wt_transform_kernel<<<(128*128*9 + 255)/256, 256, 0, stream>>>(P[3].w2, Wt, 128, 128, 9);
        conv_wmma_kernel<128,128,28,28,1,3><<<gc28, 256, 0, stream>>>(Cact, Wt, Bb);
        bn_stats_part<128><<<NPART, 256, 0, stream>>>(Bb, R28, pm, pq);
        bn_stats_final<<<1, 128, 0, stream>>>(pm, pq, 128, NPART, R28, mean2, var2);
        bn_add_relu_kernel<<<gp28, 256, 0, stream>>>(Bb, mean2, var2, P[3].g2, P[3].b2,
                                                     A, nullptr, nullptr, nullptr, nullptr,
                                                     0, o_fm[3], 128, 28 * 28, t28);
        dct_quant_wmma<28,32><<<BATCH * 128, 256, 0, stream>>>(o_fm[3], D28, o_dq[3]);
        // final reconstructed x -> first output slot (NCHW)
        idct_wmma<28,32><<<BATCH * 128, 256, 0, stream>>>(o_dq[3], Di28, nullptr, 128, out);
    }
}
